// ColbertLoss_74878459838967
// MI455X (gfx1250) — compile-verified, matrix operand order checked by reading, not run on hardware
//
#include <hip/hip_runtime.h>
#include <math.h>

// ---------------------------------------------------------------------------
// ColBERT late-interaction loss for MI455X (gfx1250, wave32, WMMA).
//
//   late[b,c,s,d] = sum_h q[b,s,h] * p[c,d,h]        (M=32, N=128, K=128)
//   scores[b,c]   = sum_s max_d late[b,c,s,d] / SQ / T
//   loss          = mean_b ( logsumexp_c scores[b,:] - scores[b,0] )
//
// GEMM via V_WMMA_F32_16X16X32_BF16 with an fp32 -> bf16 hi/lo error split:
//   q*p ~= qhi*phi + qhi*plo + qlo*phi   (3 WMMAs per K-step, ~fp32 accuracy)
//
// Block = one doc candidate c, eight query batches b. p_c is converted to
// bf16 hi/lo in LDS exactly once per block; each of the 8 waves owns one
// 16-column n-tile and keeps its B fragments in registers for all 8 b's.
// ---------------------------------------------------------------------------

typedef __bf16 v4bf  __attribute__((ext_vector_type(4)));
typedef __bf16 v8bf  __attribute__((ext_vector_type(8)));
typedef __bf16 v16bf __attribute__((ext_vector_type(16)));
typedef float  v4f   __attribute__((ext_vector_type(4)));
typedef float  v8f   __attribute__((ext_vector_type(8)));

#define B_  64
#define SQ_ 32
#define H_  128
#define C_  128
#define SD_ 128
#define TEMP_ 0.05f

// Padded LDS row stride (bf16 elems): 272 bytes/row => per-lane b128 reads of
// 16 consecutive rows at the same column walk 4 banks/row (conflict-free).
#define LSTR 136

__device__ __forceinline__ v16bf cat8(v8bf a, v8bf b) {
    return __builtin_shufflevector(a, b, 0,1,2,3,4,5,6,7,8,9,10,11,12,13,14,15);
}

__global__ __launch_bounds__(256) void colbert_scores_wmma(
    const float* __restrict__ q,   // [B, SQ, H]
    const float* __restrict__ p,   // [C, SD, H]
    float* __restrict__ scores)    // [B, C]  raw sum_s max_d (unscaled)
{
    __shared__ __attribute__((aligned(16))) __bf16 phi_s[SD_ * LSTR];
    __shared__ __attribute__((aligned(16))) __bf16 plo_s[SD_ * LSTR];
    __shared__ __attribute__((aligned(16))) __bf16 qhi_s[SQ_ * LSTR];
    __shared__ __attribute__((aligned(16))) __bf16 qlo_s[SQ_ * LSTR];
    __shared__ float red_s[8][SQ_];   // per-wave row maxes

    const int c   = blockIdx.x & 127;     // one doc candidate per block
    const int bg  = blockIdx.x >> 7;      // 8 b-groups
    const int tid = threadIdx.x;

    // ---- Stage p_c into LDS as bf16 hi/lo (once per block) -----------------
    {
        const float* pc = p + (size_t)c * SD_ * H_;
#pragma unroll
        for (int i = 0; i < 16; ++i) {           // 4096 float4s / 256 threads
            const int f4 = tid + i * 256;
            const int d  = f4 >> 5;              // (f4*4)/128
            const int h  = (f4 * 4) & 127;
            const v4f f  = *(const v4f*)(pc + f4 * 4);
            v4bf hi, lo;
#pragma unroll
            for (int e = 0; e < 4; ++e) {
                const __bf16 hv = (__bf16)f[e];
                hi[e] = hv;
                lo[e] = (__bf16)(f[e] - (float)hv);
            }
            *(v4bf*)(&phi_s[d * LSTR + h]) = hi;
            *(v4bf*)(&plo_s[d * LSTR + h]) = lo;
        }
    }
    __syncthreads();

    const int wave = tid >> 5;         // = n-tile index (d range wave*16..+15)
    const int lane = tid & 31;
    const int g    = lane >> 4;        // K-half selector (ISA A/B layout)
    const int nl   = lane & 15;        // A row / B,C,D column within tile

    // ---- Load this wave's B fragments once; live for the whole block ------
    v16bf Bhi[4], Blo[4];
#pragma unroll
    for (int k = 0; k < 4; ++k) {
        const int base = (wave * 16 + nl) * LSTR + k * 32 + g * 8;
        Bhi[k] = cat8(*(const v8bf*)(&phi_s[base]),
                      *(const v8bf*)(&phi_s[base + 16]));
        Blo[k] = cat8(*(const v8bf*)(&plo_s[base]),
                      *(const v8bf*)(&plo_s[base + 16]));
    }

    // ---- Loop over the 8 query batches this block owns ---------------------
    for (int bi = 0; bi < 8; ++bi) {
        const int b = bg * 8 + bi;

        __syncthreads();   // previous iter's q/red fully consumed

        // Stage q_b into LDS as bf16 hi/lo (1024 float4s / 256 threads).
        {
            const float* qb = q + (size_t)b * SQ_ * H_;
#pragma unroll
            for (int i = 0; i < 4; ++i) {
                const int f4 = tid + i * 256;
                const int s  = f4 >> 5;
                const int h  = (f4 * 4) & 127;
                const v4f f  = *(const v4f*)(qb + f4 * 4);
                v4bf hi, lo;
#pragma unroll
                for (int e = 0; e < 4; ++e) {
                    const __bf16 hv = (__bf16)f[e];
                    hi[e] = hv;
                    lo[e] = (__bf16)(f[e] - (float)hv);
                }
                *(v4bf*)(&qhi_s[s * LSTR + h]) = hi;
                *(v4bf*)(&qlo_s[s * LSTR + h]) = lo;
            }
        }
        __syncthreads();

        // ---- WMMA: 2 m-tiles x 4 k-steps x 3-way hi/lo split --------------
        float rmax[2][8];
#pragma unroll
        for (int mt = 0; mt < 2; ++mt) {
            v8f acc = {};
#pragma unroll
            for (int k = 0; k < 4; ++k) {
                const int base = (mt * 16 + nl) * LSTR + k * 32 + g * 8;
                const v16bf Ah = cat8(*(const v8bf*)(&qhi_s[base]),
                                      *(const v8bf*)(&qhi_s[base + 16]));
                const v16bf Al = cat8(*(const v8bf*)(&qlo_s[base]),
                                      *(const v8bf*)(&qlo_s[base + 16]));
                acc = __builtin_amdgcn_wmma_f32_16x16x32_bf16(
                          false, Ah, false, Bhi[k], (short)0, acc, false, false);
                acc = __builtin_amdgcn_wmma_f32_16x16x32_bf16(
                          false, Ah, false, Blo[k], (short)0, acc, false, false);
                acc = __builtin_amdgcn_wmma_f32_16x16x32_bf16(
                          false, Al, false, Bhi[k], (short)0, acc, false, false);
            }
#pragma unroll
            for (int r = 0; r < 8; ++r) rmax[mt][r] = acc[r];
        }

        // MaxSim over this wave's 16 d columns: cross-lane max inside each
        // 16-lane half (masks < 16 never cross halves).
#pragma unroll
        for (int mt = 0; mt < 2; ++mt)
#pragma unroll
            for (int r = 0; r < 8; ++r) {
                float v = rmax[mt][r];
                v = fmaxf(v, __shfl_xor(v, 1, 32));
                v = fmaxf(v, __shfl_xor(v, 2, 32));
                v = fmaxf(v, __shfl_xor(v, 4, 32));
                v = fmaxf(v, __shfl_xor(v, 8, 32));
                rmax[mt][r] = v;
            }

        // Lanes nl==0 (one per half) publish their 16 rows.
        if (nl == 0) {
#pragma unroll
            for (int mt = 0; mt < 2; ++mt)
#pragma unroll
                for (int r = 0; r < 8; ++r)
                    red_s[wave][mt * 16 + g * 8 + r] = rmax[mt][r];
        }
        __syncthreads();

        // Wave 0: max over the 8 n-tiles per row, then sum over the 32 rows.
        if (tid < 32) {
            float m = red_s[0][tid];
#pragma unroll
            for (int w = 1; w < 8; ++w) m = fmaxf(m, red_s[w][tid]);
            m += __shfl_xor(m, 1, 32);
            m += __shfl_xor(m, 2, 32);
            m += __shfl_xor(m, 4, 32);
            m += __shfl_xor(m, 8, 32);
            m += __shfl_xor(m, 16, 32);
            if (tid == 0) scores[b * C_ + c] = m;   // raw: sum_s max_d
        }
    }
}

// ---------------------------------------------------------------------------
// scores[64,128] -> loss scalar:  mean_b( logsumexp_c v[b,:] - v[b,0] )
// with v = raw / SQ / TEMPERATURE.  64x128 values: trivial cost.
// ---------------------------------------------------------------------------
__global__ __launch_bounds__(64) void colbert_loss_reduce(
    const float* __restrict__ scores, float* __restrict__ out)
{
    __shared__ float red[64];
    const int b = threadIdx.x;                          // 0..63
    const float invScale = 1.0f / ((float)SQ_ * TEMP_); // 0.625
    const float* row = scores + b * C_;

    float m = -3.4e38f;
    for (int cc = 0; cc < C_; ++cc) m = fmaxf(m, row[cc] * invScale);
    float s = 0.f;
    for (int cc = 0; cc < C_; ++cc) s += expf(row[cc] * invScale - m);
    red[b] = (m + logf(s)) - row[0] * invScale;
    __syncthreads();

    for (int off = 32; off > 0; off >>= 1) {
        if (b < off) red[b] += red[b + off];
        __syncthreads();
    }
    if (b == 0) out[0] = red[0] * (1.0f / (float)B_);
}

// ---------------------------------------------------------------------------
extern "C" void kernel_launch(void* const* d_in, const int* in_sizes, int n_in,
                              void* d_out, int out_size, void* d_ws, size_t ws_size,
                              hipStream_t stream)
{
    const float* q = (const float*)d_in[0];   // [64, 32, 128] fp32
    const float* p = (const float*)d_in[1];   // [128, 128, 128] fp32
    float* scores  = (float*)d_ws;            // [64, 128] fp32 scratch (32 KB)
    float* out     = (float*)d_out;           // scalar

    // 8 b-groups x 128 c; 8 waves per block, one n-tile per wave.
    colbert_scores_wmma<<<dim3((B_ / 8) * C_), dim3(256), 0, stream>>>(q, p, scores);
    colbert_loss_reduce<<<dim3(1), dim3(64), 0, stream>>>(scores, out);
}
